// GlobalAggr_59072980189219
// MI455X (gfx1250) — matrix operand order repaired and checked by reading, not compiled, not run on hardware
//
#include <hip/hip_runtime.h>
#include <hip/hip_bf16.h>
#include <math.h>

typedef __attribute__((ext_vector_type(16))) _Float16 v16h;
typedef __attribute__((ext_vector_type(8)))  float    v8f;

#define DEVINL __device__ __forceinline__

constexpr int   B_    = 4;
constexpr int   C_    = 512;
constexpr int   NTOK  = 1568;   // 8*14*14
constexpr int   NH    = 8;
constexpr int   DH    = 64;
constexpr float SCALE = 0.125f; // 64^-0.5
constexpr float EPSV  = 1e-5f;

DEVINL v8f wmma16(v16h a, v16h b, v8f c) {
  return __builtin_amdgcn_wmma_f32_16x16x32_f16(false, a, false, b, (short)0, c,
                                                false, false);
}

// A fragment: 16x32 f16 tile from row-major source, stride lda (elements).
DEVINL v16h load_a(const _Float16* A, int lda) {
  int lane = threadIdx.x & 31;
  int kb   = (lane >> 4) << 3;
  const _Float16* row = A + (size_t)(lane & 15) * lda;
  v16h a;
#pragma unroll
  for (int j = 0; j < 8; ++j) a[j]     = row[kb + j];
#pragma unroll
  for (int j = 0; j < 8; ++j) a[8 + j] = row[16 + kb + j];
  return a;
}

// B fragment: 32x16 f16 tile whose SOURCE is B^T row-major (N x K), stride ldb.
DEVINL v16h load_bt(const _Float16* BT, int ldb) {
  int lane = threadIdx.x & 31;
  const _Float16* row = BT + (size_t)(lane & 15) * ldb + ((lane >> 4) << 3);
  v16h b;
#pragma unroll
  for (int j = 0; j < 8; ++j) b[j] = row[j];
#pragma unroll
  for (int j = 0; j < 8; ++j) b[8 + j] = row[16 + j];
  return b;
}

// ---------------- kernel 1: channel norm -> XNT [b][n][c] f16 ----------------
__global__ void __launch_bounds__(256)
k_norm(const float* __restrict__ x, const float* __restrict__ nw,
       const float* __restrict__ nb, _Float16* __restrict__ xnt) {
  int wave = threadIdx.x >> 5, lane = threadIdx.x & 31;
  int gw = blockIdx.x * 8 + wave;
  int b = gw / NTOK, n = gw % NTOK;
  const float* xb = x + (size_t)b * C_ * NTOK + n;
  int c0 = lane * 16;
  float s = 0.f, ss = 0.f;
  float vals[16];
#pragma unroll
  for (int t = 0; t < 16; ++t) {
    float v = xb[(size_t)(c0 + t) * NTOK];
    vals[t] = v; s += v; ss += v * v;
  }
#pragma unroll
  for (int m = 1; m < 32; m <<= 1) {
    s  += __shfl_xor(s,  m, 32);
    ss += __shfl_xor(ss, m, 32);
  }
  float mean = s * (1.f / C_);
  float var  = ss * (1.f / C_) - mean * mean;
  float inv  = 1.f / (sqrtf(fmaxf(var, 0.f)) + EPSV);
  _Float16* xo = xnt + ((size_t)b * NTOK + n) * C_;
#pragma unroll
  for (int t = 0; t < 16; ++t) {
    int c = c0 + t;
    xo[c] = (_Float16)((vals[t] - mean) * inv * nw[c] + nb[c]);
  }
}

// ---------------- kernel 2: weight f32 -> f16 --------------------------------
__global__ void __launch_bounds__(256)
k_wconv(const float* __restrict__ wq, const float* __restrict__ wo,
        _Float16* __restrict__ wqh, _Float16* __restrict__ woh) {
  int i = blockIdx.x * 256 + threadIdx.x;
  if (i < 3 * C_ * C_) wqh[i] = (_Float16)wq[i];
  else                 woh[i - 3 * C_ * C_] = (_Float16)wo[i - 3 * C_ * C_];
}

// ---------------- kernel 3: QKV GEMM (64x32 wave tiles) + scatter ------------
__global__ void __launch_bounds__(256)
k_qkv(const _Float16* __restrict__ wq, const _Float16* __restrict__ xnt,
      _Float16* __restrict__ Q, _Float16* __restrict__ Kt,
      _Float16* __restrict__ Vt) {
  int wave = threadIdx.x >> 5, lane = threadIdx.x & 31;
  int gw = blockIdx.x * 8 + wave;             // 0 .. 4703
  int b  = gw / (24 * 49);
  int t  = gw % (24 * 49);
  int m0 = (t / 49) * 64;
  int n0 = (t % 49) * 32;
  const _Float16* Bb = xnt + (size_t)b * NTOK * C_;
  v8f acc[4][2];
#pragma unroll
  for (int mi = 0; mi < 4; ++mi) { acc[mi][0] = (v8f){}; acc[mi][1] = (v8f){}; }
  for (int kk = 0; kk < C_; kk += 32) {
    v16h B0 = load_bt(Bb + (size_t)n0 * C_ + kk, C_);
    v16h B1 = load_bt(Bb + (size_t)(n0 + 16) * C_ + kk, C_);
#pragma unroll
    for (int mi = 0; mi < 4; ++mi) {
      v16h Af = load_a(wq + (size_t)(m0 + mi * 16) * C_ + kk, C_);
      acc[mi][0] = wmma16(Af, B0, acc[mi][0]);
      acc[mi][1] = wmma16(Af, B1, acc[mi][1]);
    }
  }
  // m0 is 64-aligned -> s and h are wave-uniform
  int s  = m0 >> 9;
  int hh = (m0 >> 6) & 7;
  size_t bh = (size_t)(b * NH + hh);
  int half = lane >> 4, lc = lane & 15;
#pragma unroll
  for (int mi = 0; mi < 4; ++mi) {
#pragma unroll
    for (int r = 0; r < 8; ++r) {
      int d = mi * 16 + r + 8 * half;
#pragma unroll
      for (int ni = 0; ni < 2; ++ni) {
        int n = n0 + ni * 16 + lc;
        float val = acc[mi][ni][r];
        if (s == 0)      Q [(bh * NTOK + n) * DH + d] = (_Float16)(val * SCALE);
        else if (s == 1) Kt[(bh * NTOK + n) * DH + d] = (_Float16)val;
        else             Vt[(bh * DH + d) * NTOK + n] = (_Float16)val;
      }
    }
  }
}

// -------- async global->LDS staging of one 32-key K tile and V tile ----------
// K tile: kt rows j0..j0+31, [j][d] -> 4KB linear.  V tile: vt rows d=0..63,
// 64B segment each at column j0 -> 64 x 4 b128 transfers.  512 transfers total.
DEVINL void fill_async(const _Float16* kt, const _Float16* vt,
                       const _Float16* klds, const _Float16* vlds,
                       int j0, int tid) {
  // generic LDS pointers: low 32 bits are the LDS byte address
  unsigned kl = (unsigned)(uintptr_t)klds;
  unsigned vl = (unsigned)(uintptr_t)vlds;
  const char* kg = (const char*)(kt + (size_t)j0 * DH);
  const char* vg = (const char*)(vt + j0);
#pragma unroll
  for (int t = 0; t < 3; ++t) {
    int idx = t * 224 + tid;
    if (idx < 512) {
      unsigned lds;
      unsigned long long g;
      if (idx < 256) {                        // K tile, fully linear
        lds = kl + idx * 16;
        g   = (unsigned long long)(kg + idx * 16);
      } else {                                // V tile, 64B per d-row
        int vi = idx - 256;
        lds = vl + vi * 16;
        g   = (unsigned long long)(vg + (size_t)(vi >> 2) * (NTOK * 2) +
                                   (vi & 3) * 16);
      }
      asm volatile("global_load_async_to_lds_b128 %0, %1, off"
                   :: "v"(lds), "v"(g) : "memory");
    }
  }
}

// ---------------- kernel 4: fused flash attention (cooperative) --------------
// 224-thread block = 7 waves x 32 query rows = 224 rows of one (b,h).
// K/V tiles double-buffered in LDS via async loads, shared by all waves.
__global__ void __launch_bounds__(224)
k_attn(const _Float16* __restrict__ Q, const _Float16* __restrict__ Kt,
       const _Float16* __restrict__ Vt, _Float16* __restrict__ AOT) {
  __shared__ _Float16 ktile[2][32 * DH];      // [j][d]   4KB each
  __shared__ _Float16 vtile[2][DH * 32];      // [d][j]   4KB each
  __shared__ _Float16 pbuf[7][32 * 32];       // per-wave P staging
  int wave = threadIdx.x >> 5, lane = threadIdx.x & 31, tid = threadIdx.x;
  int bh = blockIdx.x / 7;
  int i0 = (blockIdx.x % 7) * 224 + wave * 32;
  const _Float16* q  = Q  + (size_t)bh * NTOK * DH;
  const _Float16* kt = Kt + (size_t)bh * NTOK * DH;   // [j][d]
  const _Float16* vt = Vt + (size_t)bh * DH * NTOK;   // [d][j]
  int half = lane >> 4, lc = lane & 15;

  v16h aq[2][2];
#pragma unroll
  for (int rbi = 0; rbi < 2; ++rbi)
#pragma unroll
    for (int kh = 0; kh < 2; ++kh)
      aq[rbi][kh] = load_a(q + (size_t)(i0 + rbi * 16) * DH + kh * 32, DH);

  v8f o[2][4];
  float mrow[2][8], lrow[2][8];
#pragma unroll
  for (int rbi = 0; rbi < 2; ++rbi) {
#pragma unroll
    for (int ti = 0; ti < 4; ++ti) o[rbi][ti] = (v8f){};
#pragma unroll
    for (int r = 0; r < 8; ++r) { mrow[rbi][r] = -1e30f; lrow[rbi][r] = 0.f; }
  }

  fill_async(kt, vt, &ktile[0][0], &vtile[0][0], 0, tid);

  int buf = 0;
  for (int jt = 0; jt < 49; ++jt) {
    asm volatile("s_wait_asynccnt 0x0" ::: "memory"); // my transfers done
    __syncthreads();                                  // everyone's done; prev reads done
    if (jt + 1 < 49)
      fill_async(kt, vt, &ktile[buf ^ 1][0], &vtile[buf ^ 1][0],
                 (jt + 1) * 32, tid);
    // ---- S = Q*K^T: 32 rows x 32 cols per wave, K frags shared from LDS ----
    v16h bk00 = load_bt(&ktile[buf][0], DH);
    v16h bk01 = load_bt(&ktile[buf][32], DH);
    v16h bk10 = load_bt(&ktile[buf][16 * DH], DH);
    v16h bk11 = load_bt(&ktile[buf][16 * DH + 32], DH);
#pragma unroll
    for (int rbi = 0; rbi < 2; ++rbi) {
      v8f s0 = {}, s1 = {};
      s0 = wmma16(aq[rbi][0], bk00, s0);  s0 = wmma16(aq[rbi][1], bk01, s0);
      s1 = wmma16(aq[rbi][0], bk10, s1);  s1 = wmma16(aq[rbi][1], bk11, s1);
#pragma unroll
      for (int r = 0; r < 8; ++r) {
        float tm = fmaxf(s0[r], s1[r]);
        tm = fmaxf(tm, __shfl_xor(tm, 1, 16));
        tm = fmaxf(tm, __shfl_xor(tm, 2, 16));
        tm = fmaxf(tm, __shfl_xor(tm, 4, 16));
        tm = fmaxf(tm, __shfl_xor(tm, 8, 16));
        float nm    = fmaxf(mrow[rbi][r], tm);
        float alpha = __expf(mrow[rbi][r] - nm);
        float p0 = __expf(s0[r] - nm);
        float p1 = __expf(s1[r] - nm);
        float rs = p0 + p1;
        rs += __shfl_xor(rs, 1, 16);
        rs += __shfl_xor(rs, 2, 16);
        rs += __shfl_xor(rs, 4, 16);
        rs += __shfl_xor(rs, 8, 16);
        lrow[rbi][r] = lrow[rbi][r] * alpha + rs;
        mrow[rbi][r] = nm;
        o[rbi][0][r] *= alpha; o[rbi][1][r] *= alpha;
        o[rbi][2][r] *= alpha; o[rbi][3][r] *= alpha;
        int row = rbi * 16 + r + 8 * half;
        pbuf[wave][row * 32 + lc]      = (_Float16)p0;
        pbuf[wave][row * 32 + 16 + lc] = (_Float16)p1;
      }
    }
    asm volatile("s_wait_dscnt 0x0" ::: "memory");    // in-wave P staging fence
    v16h bv0 = load_bt(&vtile[buf][0 * 16 * 32], 32);
    v16h bv1 = load_bt(&vtile[buf][1 * 16 * 32], 32);
    v16h bv2 = load_bt(&vtile[buf][2 * 16 * 32], 32);
    v16h bv3 = load_bt(&vtile[buf][3 * 16 * 32], 32);
#pragma unroll
    for (int rbi = 0; rbi < 2; ++rbi) {
      v16h ap = load_a(&pbuf[wave][rbi * 16 * 32], 32);
      o[rbi][0] = wmma16(ap, bv0, o[rbi][0]);
      o[rbi][1] = wmma16(ap, bv1, o[rbi][1]);
      o[rbi][2] = wmma16(ap, bv2, o[rbi][2]);
      o[rbi][3] = wmma16(ap, bv3, o[rbi][3]);
    }
    buf ^= 1;
  }
  // ---- normalize, store token-major [n][c] for the output projection ----
  int bb = bh >> 3, hh = bh & 7;
  _Float16* ao = AOT + ((size_t)bb * NTOK) * C_ + hh * DH;
#pragma unroll
  for (int rbi = 0; rbi < 2; ++rbi) {
#pragma unroll
    for (int r = 0; r < 8; ++r) {
      int   n   = i0 + rbi * 16 + r + 8 * half;
      float inv = 1.f / lrow[rbi][r];
      _Float16* row = ao + (size_t)n * C_;
      row[0  + lc] = (_Float16)(o[rbi][0][r] * inv);
      row[16 + lc] = (_Float16)(o[rbi][1][r] * inv);
      row[32 + lc] = (_Float16)(o[rbi][2][r] * inv);
      row[48 + lc] = (_Float16)(o[rbi][3][r] * inv);
    }
  }
}

// ---------------- kernel 5: output projection (64x32 tiles) + bias -----------
__global__ void __launch_bounds__(256)
k_out(const _Float16* __restrict__ wo, const _Float16* __restrict__ AOT,
      const float* __restrict__ bout, float* __restrict__ y) {
  int wave = threadIdx.x >> 5, lane = threadIdx.x & 31;
  int gw = blockIdx.x * 8 + wave;             // 0 .. 1567
  int b  = gw / (8 * 49);
  int t  = gw % (8 * 49);
  int m0 = (t / 49) * 64;
  int n0 = (t % 49) * 32;
  const _Float16* Bb = AOT + (size_t)b * NTOK * C_;
  v8f acc[4][2];
#pragma unroll
  for (int mi = 0; mi < 4; ++mi) { acc[mi][0] = (v8f){}; acc[mi][1] = (v8f){}; }
  for (int kk = 0; kk < C_; kk += 32) {
    v16h B0 = load_bt(Bb + (size_t)n0 * C_ + kk, C_);
    v16h B1 = load_bt(Bb + (size_t)(n0 + 16) * C_ + kk, C_);
#pragma unroll
    for (int mi = 0; mi < 4; ++mi) {
      v16h Af = load_a(wo + (size_t)(m0 + mi * 16) * C_ + kk, C_);
      acc[mi][0] = wmma16(Af, B0, acc[mi][0]);
      acc[mi][1] = wmma16(Af, B1, acc[mi][1]);
    }
  }
  int half = lane >> 4, lc = lane & 15;
#pragma unroll
  for (int mi = 0; mi < 4; ++mi) {
#pragma unroll
    for (int r = 0; r < 8; ++r) {
      int o = m0 + mi * 16 + r + 8 * half;
      float bb = bout[o];
#pragma unroll
      for (int ni = 0; ni < 2; ++ni) {
        int n = n0 + ni * 16 + lc;
        y[((size_t)b * C_ + o) * NTOK + n] = acc[mi][ni][r] + bb;
      }
    }
  }
}

// ---------------- host side --------------------------------------------------
extern "C" void kernel_launch(void* const* d_in, const int* in_sizes, int n_in,
                              void* d_out, int out_size, void* d_ws, size_t ws_size,
                              hipStream_t stream) {
  const float* x  = (const float*)d_in[0];
  const float* nw = (const float*)d_in[1];
  const float* nb = (const float*)d_in[2];
  const float* wq = (const float*)d_in[3];
  const float* wo = (const float*)d_in[4];
  const float* bo = (const float*)d_in[5];
  float* y = (float*)d_out;

  char*  ws  = (char*)d_ws;
  size_t off = 0;
  auto alloc = [&](size_t bytes) -> char* {
    char* p = ws + off;
    off += (bytes + 255) & ~(size_t)255;
    return p;
  };
  _Float16* XNT = (_Float16*)alloc((size_t)B_ * NTOK * C_ * 2);
  _Float16* WQ  = (_Float16*)alloc((size_t)3 * C_ * C_ * 2);
  _Float16* WO  = (_Float16*)alloc((size_t)C_ * C_ * 2);
  _Float16* Qs  = (_Float16*)alloc((size_t)B_ * NH * NTOK * DH * 2);
  _Float16* Ks  = (_Float16*)alloc((size_t)B_ * NH * NTOK * DH * 2);
  _Float16* Vs  = (_Float16*)alloc((size_t)B_ * NH * NTOK * DH * 2);
  _Float16* AOT = (_Float16*)alloc((size_t)B_ * NTOK * C_ * 2);
  (void)ws_size; (void)in_sizes; (void)n_in; (void)out_size;

  k_norm <<<784,  256, 0, stream>>>(x, nw, nb, XNT);
  k_wconv<<<4096, 256, 0, stream>>>(wq, wo, WQ, WO);
  k_qkv  <<<588,  256, 0, stream>>>(WQ, XNT, Qs, Ks, Vs);
  k_attn <<<224,  224, 0, stream>>>(Qs, Ks, Vs, AOT);
  k_out  <<<196,  256, 0, stream>>>(WO, AOT, bo, y);
}
